// CrossViewTransformer_1159641170455
// MI455X (gfx1250) — matrix operand order repaired and checked by reading, not compiled, not run on hardware
//
#include <hip/hip_runtime.h>

typedef __attribute__((ext_vector_type(2)))  float    v2f;
typedef __attribute__((ext_vector_type(8)))  float    v8f;
typedef __attribute__((ext_vector_type(4)))  float    v4f;
typedef __attribute__((ext_vector_type(16))) _Float16 v16h;
typedef __attribute__((ext_vector_type(8)))  _Float16 v8h;

#define BATCH 2
#define CH    64
#define DH    8
#define NPIX  9216          // 96*96
#define JTILES_PER_B (NPIX / 16)   // 576
#define NTILES (BATCH * JTILES_PER_B) // 1152

// ---------------------------------------------------------------------------
// Kernel 1: fused 1x1 convs -> Af (k-features, [B][N][8] f32),
//                              Bf (q-features, [B][N][8] f32),
//                              Vf (v-features, [B][64][N] f16, channel-major)
// ---------------------------------------------------------------------------
__global__ __launch_bounds__(256) void qkv_kernel(
    const float* __restrict__ rv, const float* __restrict__ bev,
    const float* __restrict__ Wq, const float* __restrict__ bq,
    const float* __restrict__ Wk, const float* __restrict__ bk,
    const float* __restrict__ Wv, const float* __restrict__ bv,
    float* __restrict__ Af, float* __restrict__ Bf, _Float16* __restrict__ Vf)
{
    int t = blockIdx.x * blockDim.x + threadIdx.x;
    if (t >= BATCH * NPIX) return;
    int b = t / NPIX;
    int j = t - b * NPIX;

    const float* rvp  = rv  + (size_t)b * CH * NPIX + j;
    const float* bevp = bev + (size_t)b * CH * NPIX + j;

    float ka[DH], qa[DH], va[CH];
#pragma unroll
    for (int d = 0; d < DH; ++d) { ka[d] = bk[d]; qa[d] = bq[d]; }
#pragma unroll 8
    for (int c = 0; c < CH; ++c) va[c] = bv[c];

    for (int c = 0; c < CH; ++c) {
        float xr = rvp[(size_t)c * NPIX];
        float xb = bevp[(size_t)c * NPIX];
#pragma unroll
        for (int d = 0; d < DH; ++d) {
            ka[d] = fmaf(Wk[d * CH + c], xr, ka[d]);
            qa[d] = fmaf(Wq[d * CH + c], xb, qa[d]);
        }
#pragma unroll 8
        for (int o = 0; o < CH; ++o)
            va[o] = fmaf(Wv[o * CH + c], xr, va[o]);
    }

    float* afp = Af + ((size_t)b * NPIX + j) * DH;
    float* bfp = Bf + ((size_t)b * NPIX + j) * DH;
#pragma unroll
    for (int d = 0; d < DH; ++d) { afp[d] = ka[d]; bfp[d] = qa[d]; }
    _Float16* vfp = Vf + (size_t)b * CH * NPIX + j;
#pragma unroll 8
    for (int o = 0; o < CH; ++o) vfp[(size_t)o * NPIX] = (_Float16)va[o];
}

// ---------------------------------------------------------------------------
// Kernel 2: fused flash attention, one wave32 per 16-row output tile,
// 64 keys per iteration (8x v_wmma_f32_16x16x4_f32 + 8x v_wmma_f32_16x16x32_f16).
// S^T D-layout (lane=j-col, vgpr=i-row+8*half) == f16 A-operand layout for P,
// so the softmax->P path is register-resident (no LDS transpose).
// ---------------------------------------------------------------------------
__global__ __launch_bounds__(256) void attn_kernel(
    const float* __restrict__ Af, const float* __restrict__ Bf,
    const _Float16* __restrict__ Vf, const float* __restrict__ bev,
    float* __restrict__ out)
{
    int wave = (int)((blockIdx.x * blockDim.x + threadIdx.x) >> 5);
    int lane = (int)(threadIdx.x & 31);
    int hlf  = lane >> 4;
    int l15  = lane & 15;
    if (wave >= NTILES) return;

    int b     = wave / JTILES_PER_B;
    int jbase = (wave - b * JTILES_PER_B) * 16;

    const float*    AfB = Af + (size_t)b * NPIX * DH;
    const float*    BfB = Bf + (size_t)b * NPIX * DH;
    const _Float16* VfB = Vf + (size_t)b * CH * NPIX;

    // Loop-invariant B-operands (k-features, column j = jbase + l15).
    // f32 16x16x4 B layout: lanes0-15 supply K=0,1 ; lanes16-31 supply K=2,3.
    const float* afj = AfB + (size_t)(jbase + l15) * DH;
    v2f bop0 = *(const v2f*)(afj + 2 * hlf);        // d 0..3 slice
    v2f bop1 = *(const v2f*)(afj + 4 + 2 * hlf);    // d 4..7 slice

    v8f o0 = {}, o1 = {}, o2 = {}, o3 = {};
    float rowm = -1e30f, rowl = 0.0f;

    for (int i0 = 0; i0 < NPIX; i0 += 64) {
        // ---- S^T for 4 sub-tiles of 16 keys each ----
        v8f s[4];
#pragma unroll
        for (int tsub = 0; tsub < 4; ++tsub) {
            const float* bfi = BfB + (size_t)(i0 + tsub * 16 + l15) * DH;
            v2f a0 = *(const v2f*)(bfi + 2 * hlf);
            v2f a1 = *(const v2f*)(bfi + 4 + 2 * hlf);
            v8f acc = {};
            acc = __builtin_amdgcn_wmma_f32_16x16x4_f32(false, a0, false, bop0,
                                                        (short)0, acc, false, false);
            acc = __builtin_amdgcn_wmma_f32_16x16x4_f32(false, a1, false, bop1,
                                                        (short)0, acc, false, false);
            s[tsub] = acc;
        }
        // lane holds S^T row j = jbase+l15, i = i0 + 16*tsub + r + 8*hlf

        // ---- chunk max (32 values per lane), merge lane halves ----
        float cmax = s[0][0];
#pragma unroll
        for (int tsub = 0; tsub < 4; ++tsub)
#pragma unroll
            for (int r = 0; r < 8; ++r) cmax = fmaxf(cmax, s[tsub][r]);
        cmax = fmaxf(cmax, __shfl_xor(cmax, 16, 32));

        // ---- conditional accumulator rescale (rarely taken after warmup) ----
        if (__any(cmax > rowm)) {
            float newm = fmaxf(rowm, cmax);
            float sc0  = __expf(rowm - newm);
            rowm = newm;
            rowl *= sc0;
#pragma unroll
            for (int r = 0; r < 8; ++r) {
                float sc = __shfl(sc0, r + 8 * hlf, 32);   // row r + 8*hlf
                o0[r] *= sc; o1[r] *= sc; o2[r] *= sc; o3[r] *= sc;
            }
        }

        // ---- exp + pack P into two f16 A-operands (identity lane mapping) ----
        float psum = 0.0f;
        v16h pa0, pa1;
#pragma unroll
        for (int r = 0; r < 8; ++r) {
            float e0 = __expf(s[0][r] - rowm);
            float e1 = __expf(s[1][r] - rowm);
            float e2 = __expf(s[2][r] - rowm);
            float e3 = __expf(s[3][r] - rowm);
            psum += (e0 + e1) + (e2 + e3);
            pa0[r] = (_Float16)e0; pa0[8 + r] = (_Float16)e1;
            pa1[r] = (_Float16)e2; pa1[8 + r] = (_Float16)e3;
        }
        psum += __shfl_xor(psum, 16, 32);
        rowl += psum;

        // ---- P(16x64) x V(64x64): 4 channel tiles x 2 K-chunks ----
        // f16 B layout: lanes0-15 hold K=0..7,16..23, lanes16-31 hold K=8..15,24..31.
        const _Float16* vrow = VfB + (size_t)l15 * NPIX + (i0 + 8 * hlf);
        v16h vb0 = __builtin_shufflevector(*(const v8h*)(vrow + (size_t)0  * NPIX), *(const v8h*)(vrow + 16 + (size_t)0  * NPIX), 0,1,2,3,4,5,6,7,8,9,10,11,12,13,14,15);
        v16h vb1 = __builtin_shufflevector(*(const v8h*)(vrow + (size_t)16 * NPIX), *(const v8h*)(vrow + 16 + (size_t)16 * NPIX), 0,1,2,3,4,5,6,7,8,9,10,11,12,13,14,15);
        v16h vb2 = __builtin_shufflevector(*(const v8h*)(vrow + (size_t)32 * NPIX), *(const v8h*)(vrow + 16 + (size_t)32 * NPIX), 0,1,2,3,4,5,6,7,8,9,10,11,12,13,14,15);
        v16h vb3 = __builtin_shufflevector(*(const v8h*)(vrow + (size_t)48 * NPIX), *(const v8h*)(vrow + 16 + (size_t)48 * NPIX), 0,1,2,3,4,5,6,7,8,9,10,11,12,13,14,15);
        o0 = __builtin_amdgcn_wmma_f32_16x16x32_f16(false, pa0, false, vb0, (short)0, o0, false, false);
        o1 = __builtin_amdgcn_wmma_f32_16x16x32_f16(false, pa0, false, vb1, (short)0, o1, false, false);
        o2 = __builtin_amdgcn_wmma_f32_16x16x32_f16(false, pa0, false, vb2, (short)0, o2, false, false);
        o3 = __builtin_amdgcn_wmma_f32_16x16x32_f16(false, pa0, false, vb3, (short)0, o3, false, false);
        v16h vc0 = __builtin_shufflevector(*(const v8h*)(vrow + 32 + (size_t)0  * NPIX), *(const v8h*)(vrow + 48 + (size_t)0  * NPIX), 0,1,2,3,4,5,6,7,8,9,10,11,12,13,14,15);
        v16h vc1 = __builtin_shufflevector(*(const v8h*)(vrow + 32 + (size_t)16 * NPIX), *(const v8h*)(vrow + 48 + (size_t)16 * NPIX), 0,1,2,3,4,5,6,7,8,9,10,11,12,13,14,15);
        v16h vc2 = __builtin_shufflevector(*(const v8h*)(vrow + 32 + (size_t)32 * NPIX), *(const v8h*)(vrow + 48 + (size_t)32 * NPIX), 0,1,2,3,4,5,6,7,8,9,10,11,12,13,14,15);
        v16h vc3 = __builtin_shufflevector(*(const v8h*)(vrow + 32 + (size_t)48 * NPIX), *(const v8h*)(vrow + 48 + (size_t)48 * NPIX), 0,1,2,3,4,5,6,7,8,9,10,11,12,13,14,15);
        o0 = __builtin_amdgcn_wmma_f32_16x16x32_f16(false, pa1, false, vc0, (short)0, o0, false, false);
        o1 = __builtin_amdgcn_wmma_f32_16x16x32_f16(false, pa1, false, vc1, (short)0, o1, false, false);
        o2 = __builtin_amdgcn_wmma_f32_16x16x32_f16(false, pa1, false, vc2, (short)0, o2, false, false);
        o3 = __builtin_amdgcn_wmma_f32_16x16x32_f16(false, pa1, false, vc3, (short)0, o3, false, false);
    }

    // ---- epilogue: out = bev_x + O / l ----
    float linv = 1.0f / rowl;   // one reciprocal per row, broadcast below
    const float* bevB = bev + (size_t)b * CH * NPIX;
    float*       outB = out + (size_t)b * CH * NPIX;
    int joff = jbase + 8 * hlf;
#pragma unroll
    for (int ct = 0; ct < 4; ++ct) {
        v8f oc = (ct == 0) ? o0 : (ct == 1) ? o1 : (ct == 2) ? o2 : o3;
        size_t base = (size_t)(ct * 16 + l15) * NPIX + joff;
        v4f lo, hi;
#pragma unroll
        for (int r = 0; r < 8; ++r) {
            float lr  = __shfl(linv, r + 8 * hlf, 32);
            float val = fmaf(oc[r], lr, bevB[base + r]);
            if (r < 4) lo[r] = val; else hi[r - 4] = val;
        }
        *(v4f*)(outB + base)     = lo;
        *(v4f*)(outB + base + 4) = hi;
    }
}

// ---------------------------------------------------------------------------
extern "C" void kernel_launch(void* const* d_in, const int* in_sizes, int n_in,
                              void* d_out, int out_size, void* d_ws, size_t ws_size,
                              hipStream_t stream) {
    const float* rv  = (const float*)d_in[0];
    const float* bev = (const float*)d_in[1];
    const float* Wq  = (const float*)d_in[2];
    const float* bq  = (const float*)d_in[3];
    const float* Wk  = (const float*)d_in[4];
    const float* bk  = (const float*)d_in[5];
    const float* Wv  = (const float*)d_in[6];
    const float* bv  = (const float*)d_in[7];
    float* out = (float*)d_out;

    size_t afBytes = (size_t)BATCH * NPIX * DH * sizeof(float);   // 576 KB
    float*    Af = (float*)d_ws;
    float*    Bf = (float*)((char*)d_ws + afBytes);
    _Float16* Vf = (_Float16*)((char*)d_ws + 2 * afBytes);        // +2.25 MB

    int nthreads = BATCH * NPIX;
    qkv_kernel<<<(nthreads + 255) / 256, 256, 0, stream>>>(
        rv, bev, Wq, bq, Wk, bk, Wv, bv, Af, Bf, Vf);

    attn_kernel<<<(NTILES * 32 + 255) / 256, 256, 0, stream>>>(
        Af, Bf, Vf, bev, out);
}
